// DetrAttention_64665027608812
// MI455X (gfx1250) — matrix operand order repaired and checked by reading, not compiled, not run on hardware
//
#include <hip/hip_runtime.h>

typedef _Float16 half8 __attribute__((ext_vector_type(8)));
typedef _Float16 v16h  __attribute__((ext_vector_type(16)));
typedef float    v8f   __attribute__((ext_vector_type(8)));

union F16x16 { v16h v; half8 h8[2]; };

#define S_LEN 1024
#define D_EMB 256
#define NHEAD 8
#define DHEAD 32

__device__ __forceinline__ v8f wmma_f16(v16h a, v16h b, v8f c) {
  // 8 args: (neg_a, A, neg_b, B, c_mod, C, reuse_a, reuse_b)
  return __builtin_amdgcn_wmma_f32_16x16x32_f16(false, a, false, b, (short)0, c,
                                                false, false);
}

__global__ void prep_add_f16(const float* __restrict__ hs, const float* __restrict__ oq,
                             _Float16* __restrict__ x1, _Float16* __restrict__ x0, int n) {
  int i = blockIdx.x * blockDim.x + threadIdx.x;
  if (i < n) {
    float h = hs[i];
    x1[i] = (_Float16)(h + oq[i]);
    x0[i] = (_Float16)h;
  }
}

__global__ void conv_f16(const float* __restrict__ src, _Float16* __restrict__ dst, int n) {
  int i = blockIdx.x * blockDim.x + threadIdx.x;
  if (i < n) dst[i] = (_Float16)src[i];
}

// Y = (X @ W^T + bias) * scale ; X:[M,256] f16, W:[256,256] f16 (nn.Linear layout)
// Each wave computes a 32x64 register tile: 2 A-frags x 4 B-frags, 8 accumulators.
// mode 0: out16 row-major [M,256]
// mode 1: out16 transposed-V layout [B,H,Dh,S]
// mode 2: out32 row-major [M,256] (final output)
__global__ __launch_bounds__(128) void gemm_wmma(
    const _Float16* __restrict__ X, const _Float16* __restrict__ W,
    const float* __restrict__ bias, float scale, int M,
    _Float16* __restrict__ out16, float* __restrict__ out32, int mode) {
  const int lane  = threadIdx.x & 31;
  const int wave  = threadIdx.x >> 5;
  const int half_ = lane >> 4;
  const int r     = lane & 15;

  const int tiles_n = D_EMB / 64;  // 4 column tiles of 64
  int tile = blockIdx.x * 4 + wave;
  int tm = tile / tiles_n;
  int tn = tile % tiles_n;
  if (tm * 32 >= M) return;

  const _Float16* arow0 = X + (size_t)(tm * 32 + r) * D_EMB;
  const _Float16* arow1 = X + (size_t)(tm * 32 + 16 + r) * D_EMB;

  v8f acc[2][4];
#pragma unroll
  for (int i = 0; i < 2; ++i)
#pragma unroll
    for (int j = 0; j < 4; ++j) acc[i][j] = v8f{};

#pragma unroll
  for (int kb = 0; kb < D_EMB; kb += 32) {
    F16x16 a0, a1;
    // 16-bit A layout: elems 0..7 = K[kb+8*half ..], elems 8..15 = K[kb+16+8*half ..]
    a0.h8[0] = *(const half8*)(arow0 + kb + half_ * 8);
    a0.h8[1] = *(const half8*)(arow0 + kb + 16 + half_ * 8);
    a1.h8[0] = *(const half8*)(arow1 + kb + half_ * 8);
    a1.h8[1] = *(const half8*)(arow1 + kb + 16 + half_ * 8);
#pragma unroll
    for (int nb = 0; nb < 4; ++nb) {
      // 16-bit B layout: elems 0..7 = K[kb+8*half ..], elems 8..15 = K[kb+16+8*half ..], lane = out column
      const _Float16* brow = W + (size_t)(tn * 64 + nb * 16 + r) * D_EMB + kb;
      F16x16 b;
      b.h8[0] = *(const half8*)(brow + half_ * 8);
      b.h8[1] = *(const half8*)(brow + 16 + half_ * 8);
      acc[0][nb] = wmma_f16(a0.v, b.v, acc[0][nb]);
      acc[1][nb] = wmma_f16(a1.v, b.v, acc[1][nb]);
    }
  }

#pragma unroll
  for (int i = 0; i < 2; ++i) {
#pragma unroll
    for (int nb = 0; nb < 4; ++nb) {
      int col = tn * 64 + nb * 16 + r;
      float bv = bias[col];
#pragma unroll
      for (int rr = 0; rr < 8; ++rr) {
        int row = tm * 32 + i * 16 + rr + 8 * half_;  // C layout: VGPR rr -> row rr+8*half
        float val = (acc[i][nb][rr] + bv) * scale;
        if (mode == 0) {
          out16[(size_t)row * D_EMB + col] = (_Float16)val;
        } else if (mode == 1) {
          int bb = row >> 10, s = row & (S_LEN - 1);
          int hh = col >> 5, dd = col & (DHEAD - 1);
          out16[(((size_t)(bb * NHEAD + hh) * DHEAD) + dd) * S_LEN + s] = (_Float16)val;
        } else {
          out32[(size_t)row * D_EMB + col] = val;
        }
      }
    }
  }
}

// Flash attention: one wave per 32-row q-tile of one (b,h).
// Per 32-key chunk: 4 QK^T WMMAs + 4 PV WMMAs against one set of K/V fragment loads.
__global__ __launch_bounds__(128) void attn_wmma(
    const _Float16* __restrict__ Q, const _Float16* __restrict__ K,
    const _Float16* __restrict__ VT, _Float16* __restrict__ CTX) {
  __shared__ __align__(16) _Float16 plds[4][32 * 32];
  const int lane  = threadIdx.x & 31;
  const int wave  = threadIdx.x >> 5;
  const int half_ = lane >> 4;
  const int r     = lane & 15;

  int gwave = blockIdx.x * 4 + wave;
  int qt = gwave & 31;  // 32 q-tiles (of 32 rows) per (b,h)
  int bh = gwave >> 5;
  int b = bh >> 3;
  int h = bh & 7;

  const _Float16* Qb  = Q  + (size_t)(b * S_LEN + qt * 32) * D_EMB + h * DHEAD;
  const _Float16* Kb  = K  + (size_t)b * S_LEN * D_EMB + h * DHEAD;
  const _Float16* VTb = VT + (size_t)(b * NHEAD + h) * DHEAD * S_LEN;
  _Float16*       Cb  = CTX + (size_t)(b * S_LEN + qt * 32) * D_EMB + h * DHEAD;

  // Two Q A-fragments (rows 0..15 and 16..31 of the tile); Dh == WMMA K == 32
  F16x16 qa0, qa1;
  {
    const _Float16* q0 = Qb + (size_t)r * D_EMB;
    qa0.h8[0] = *(const half8*)(q0 + half_ * 8);
    qa0.h8[1] = *(const half8*)(q0 + 16 + half_ * 8);
    const _Float16* q1 = Qb + (size_t)(16 + r) * D_EMB;
    qa1.h8[0] = *(const half8*)(q1 + half_ * 8);
    qa1.h8[1] = *(const half8*)(q1 + 16 + half_ * 8);
  }

  v8f m0, m1, l0, l1, o00, o01, o10, o11;
#pragma unroll
  for (int e = 0; e < 8; ++e) {
    m0[e] = -1e30f; m1[e] = -1e30f; l0[e] = 0.f; l1[e] = 0.f;
    o00[e] = 0.f; o01[e] = 0.f; o10[e] = 0.f; o11[e] = 0.f;
  }

  for (int kb = 0; kb < S_LEN; kb += 32) {
    // Prefetch next chunk of K and VT into cache (global_prefetch_b8)
    if (kb + 32 < S_LEN) {
      __builtin_prefetch(Kb + (size_t)(kb + 32 + r + half_ * 16) * D_EMB, 0, 1);
      __builtin_prefetch(VTb + (size_t)(r + half_ * 16) * S_LEN + kb + 32, 0, 1);
    }

    // K chunk as two B-fragments: lane = key, elems 0..7 = Dh[8*half ..], 8..15 = Dh[16+8*half ..]
    F16x16 bk0, bk1;
    {
      const _Float16* k0 = Kb + (size_t)(kb + r) * D_EMB;
      bk0.h8[0] = *(const half8*)(k0 + half_ * 8);
      bk0.h8[1] = *(const half8*)(k0 + 16 + half_ * 8);
      const _Float16* k1 = Kb + (size_t)(kb + 16 + r) * D_EMB;
      bk1.h8[0] = *(const half8*)(k1 + half_ * 8);
      bk1.h8[1] = *(const half8*)(k1 + 16 + half_ * 8);
    }
    v8f zero = {};
    v8f s00 = wmma_f16(qa0.v, bk0.v, zero);  // rows 0-15, keys kb..+15
    v8f s01 = wmma_f16(qa0.v, bk1.v, zero);  // rows 0-15, keys kb+16..+31
    v8f s10 = wmma_f16(qa1.v, bk0.v, zero);  // rows 16-31, keys kb..+15
    v8f s11 = wmma_f16(qa1.v, bk1.v, zero);  // rows 16-31, keys kb+16..+31

    // per-row chunk max (reduce across the 16 lanes of each half)
    v8f cm0, cm1;
#pragma unroll
    for (int e = 0; e < 8; ++e) {
      cm0[e] = fmaxf(s00[e], s01[e]);
      cm1[e] = fmaxf(s10[e], s11[e]);
    }
#pragma unroll
    for (int off = 8; off >= 1; off >>= 1) {
#pragma unroll
      for (int e = 0; e < 8; ++e) {
        cm0[e] = fmaxf(cm0[e], __shfl_xor(cm0[e], off, 32));
        cm1[e] = fmaxf(cm1[e], __shfl_xor(cm1[e], off, 32));
      }
    }

    v8f p00, p01, p10, p11, rs0, rs1;
#pragma unroll
    for (int e = 0; e < 8; ++e) {
      float mn0 = fmaxf(m0[e], cm0[e]);
      float mn1 = fmaxf(m1[e], cm1[e]);
      float sc0 = __expf(m0[e] - mn0);
      float sc1 = __expf(m1[e] - mn1);
      p00[e] = __expf(s00[e] - mn0);
      p01[e] = __expf(s01[e] - mn0);
      p10[e] = __expf(s10[e] - mn1);
      p11[e] = __expf(s11[e] - mn1);
      rs0[e] = p00[e] + p01[e];
      rs1[e] = p10[e] + p11[e];
      o00[e] *= sc0; o01[e] *= sc0;
      o10[e] *= sc1; o11[e] *= sc1;
      l0[e] = l0[e] * sc0;
      l1[e] = l1[e] * sc1;
      m0[e] = mn0; m1[e] = mn1;
    }
#pragma unroll
    for (int off = 8; off >= 1; off >>= 1) {
#pragma unroll
      for (int e = 0; e < 8; ++e) {
        rs0[e] += __shfl_xor(rs0[e], off, 32);
        rs1[e] += __shfl_xor(rs1[e], off, 32);
      }
    }
#pragma unroll
    for (int e = 0; e < 8; ++e) { l0[e] += rs0[e]; l1[e] += rs1[e]; }

    // P (32 rows x 32 keys) C-layout -> LDS -> A-layout
    _Float16* pl = plds[wave];
#pragma unroll
    for (int rr = 0; rr < 8; ++rr) {
      int row = rr + 8 * half_;
      pl[row * 32 + r]             = (_Float16)p00[rr];
      pl[row * 32 + 16 + r]        = (_Float16)p01[rr];
      pl[(16 + row) * 32 + r]      = (_Float16)p10[rr];
      pl[(16 + row) * 32 + 16 + r] = (_Float16)p11[rr];
    }
    __builtin_amdgcn_wave_barrier();
    F16x16 pa0, pa1;
    pa0.h8[0] = *(const half8*)(pl + r * 32 + half_ * 8);
    pa0.h8[1] = *(const half8*)(pl + r * 32 + 16 + half_ * 8);
    pa1.h8[0] = *(const half8*)(pl + (16 + r) * 32 + half_ * 8);
    pa1.h8[1] = *(const half8*)(pl + (16 + r) * 32 + 16 + half_ * 8);
    __builtin_amdgcn_wave_barrier();

    // V chunk as B-fragments from pre-transposed VT: lane = dim, elems keys 8*half.., 16+8*half..
    F16x16 bv0, bv1;
    {
      const _Float16* v0 = VTb + (size_t)r * S_LEN + kb + half_ * 8;
      bv0.h8[0] = *(const half8*)(v0);
      bv0.h8[1] = *(const half8*)(v0 + 16);
      const _Float16* v1 = VTb + (size_t)(16 + r) * S_LEN + kb + half_ * 8;
      bv1.h8[0] = *(const half8*)(v1);
      bv1.h8[1] = *(const half8*)(v1 + 16);
    }
    o00 = wmma_f16(pa0.v, bv0.v, o00);  // rows 0-15,  dims 0..15
    o01 = wmma_f16(pa0.v, bv1.v, o01);  // rows 0-15,  dims 16..31
    o10 = wmma_f16(pa1.v, bv0.v, o10);  // rows 16-31, dims 0..15
    o11 = wmma_f16(pa1.v, bv1.v, o11);  // rows 16-31, dims 16..31
  }

  // normalize and store context (f16)
#pragma unroll
  for (int rr = 0; rr < 8; ++rr) {
    float inv0 = 1.0f / l0[rr];
    float inv1 = 1.0f / l1[rr];
    int row = rr + 8 * half_;
    Cb[(size_t)row * D_EMB + r]             = (_Float16)(o00[rr] * inv0);
    Cb[(size_t)row * D_EMB + 16 + r]        = (_Float16)(o01[rr] * inv0);
    Cb[(size_t)(16 + row) * D_EMB + r]      = (_Float16)(o10[rr] * inv1);
    Cb[(size_t)(16 + row) * D_EMB + 16 + r] = (_Float16)(o11[rr] * inv1);
  }
}

extern "C" void kernel_launch(void* const* d_in, const int* in_sizes, int n_in,
                              void* d_out, int out_size, void* d_ws, size_t ws_size,
                              hipStream_t stream) {
  const float* hs = (const float*)d_in[0];
  const float* oq = (const float*)d_in[1];
  const float* Wq = (const float*)d_in[2];
  const float* bq = (const float*)d_in[3];
  const float* Wk = (const float*)d_in[4];
  const float* bk = (const float*)d_in[5];
  const float* Wv = (const float*)d_in[6];
  const float* bv = (const float*)d_in[7];
  const float* Wo = (const float*)d_in[8];
  const float* bo = (const float*)d_in[9];
  float* out = (float*)d_out;

  char* ws = (char*)d_ws;
  const size_t NTOK = 8 * 1024;                        // B*S
  const size_t XSZ  = NTOK * 256 * sizeof(_Float16);   // 4 MiB
  const size_t WSZ  = 256 * 256 * sizeof(_Float16);    // 128 KiB
  _Float16* X1   = (_Float16*)(ws);                    // f16(hs + oq)
  _Float16* X0   = (_Float16*)(ws + XSZ);              // f16(hs); reused for CTX later
  _Float16* Wq16 = (_Float16*)(ws + 2 * XSZ);
  _Float16* Wk16 = (_Float16*)(ws + 2 * XSZ + WSZ);
  _Float16* Wv16 = (_Float16*)(ws + 2 * XSZ + 2 * WSZ);
  _Float16* Wo16 = (_Float16*)(ws + 2 * XSZ + 3 * WSZ);
  _Float16* Q16  = (_Float16*)(ws + 2 * XSZ + 4 * WSZ);
  _Float16* K16  = (_Float16*)(ws + 3 * XSZ + 4 * WSZ);
  _Float16* VT16 = (_Float16*)(ws + 4 * XSZ + 4 * WSZ);
  _Float16* C16  = X0;  // hs-f16 is dead after the V projection; reuse for context

  int n = (int)(NTOK * 256);
  prep_add_f16<<<(n + 255) / 256, 256, 0, stream>>>(hs, oq, X1, X0, n);
  conv_f16<<<256, 256, 0, stream>>>(Wq, Wq16, 65536);
  conv_f16<<<256, 256, 0, stream>>>(Wk, Wk16, 65536);
  conv_f16<<<256, 256, 0, stream>>>(Wv, Wv16, 65536);
  conv_f16<<<256, 256, 0, stream>>>(Wo, Wo16, 65536);

  const int M = (int)NTOK;
  // 32x64 tiles per wave: (M/32) * (256/64) = 1024 wave-tiles, 4 waves/block
  const int gemm_blocks = (M / 32) * (D_EMB / 64) / 4;  // 256
  // q = (x1 @ Wq^T + bq) * sqrt(Dh)
  gemm_wmma<<<gemm_blocks, 128, 0, stream>>>(X1, Wq16, bq, 5.656854249492381f, M, Q16, nullptr, 0);
  gemm_wmma<<<gemm_blocks, 128, 0, stream>>>(X1, Wk16, bk, 1.0f, M, K16, nullptr, 0);
  gemm_wmma<<<gemm_blocks, 128, 0, stream>>>(X0, Wv16, bv, 1.0f, M, VT16, nullptr, 1);

  // flash attention: B*H*(S/32) = 2048 waves, 4 per block
  attn_wmma<<<512, 128, 0, stream>>>(Q16, K16, VT16, C16);

  // out = ctx @ Wo^T + bo (fp32 output)
  gemm_wmma<<<gemm_blocks, 128, 0, stream>>>(C16, Wo16, bo, 1.0f, M, nullptr, out, 2);
}
